// QuadNeighborhoodEncoderAttention_7670811591055
// MI455X (gfx1250) — matrix operand order repaired and checked by reading, not compiled, not run on hardware
//
#include <hip/hip_runtime.h>

// MI455X / gfx1250, wave32. Compute-bound MLP-attention (~80 GFLOP vs ~52MB I/O):
// all GEMMs via v_wmma_f32_16x16x32_f16, f16 weights in LDS, f32 accumulate,
// fast tanh via v_exp_f32 + v_rcp_f32.
// - One-time prep kernel converts weights to transposed f16 fragment layout.
// - Main kernels: 64 samples/block = 384 rows = 24 waves (768 thr), ~300KB LDS,
//   one workgroup per 320KB WGP -> 6 waves/SIMD32.

typedef _Float16 half_t;
typedef __attribute__((ext_vector_type(16))) _Float16 v16h;
typedef __attribute__((ext_vector_type(8)))  _Float16 v8h;
typedef __attribute__((ext_vector_type(8)))  float    v8f;

#define SELF_DIM 18
#define NBR_DIM  6
#define KNBR     6
#define HDIM     128
#define OBS_DIM  54      // SELF_DIM + KNBR*NBR_DIM
#define IN1PAD   32      // 24 padded to 32 for one f16 WMMA k-chunk
#define SPB      64      // samples per block
#define RPB      (SPB*KNBR)  // 384 rows per block (24 tiles of 16)
#define NTHREADS 768     // 24 waves, one 16-row tile each

// ---- fast tanh: exp(-2|x|) via v_exp_f32, then v_rcp_f32 ----
__device__ __forceinline__ float ftanh(float x) {
  float ax = __builtin_fabsf(x);
  float t  = __expf(-2.0f * ax);
  float r  = (1.0f - t) * __builtin_amdgcn_rcpf(1.0f + t);
  return __builtin_copysignf(r, x);
}

// ---- WMMA fragment loaders ----
// A 16x32: lane<16 -> row M=lane, K {0..7,16..23}; lane>=16 -> row M=lane-16, K {8..15,24..31}
__device__ __forceinline__ v16h load_fragA(const half_t* buf, int stride, int row0, int k0, int lane) {
  const int sel = lane >> 4;
  const int r   = row0 + (lane & 15);
  const half_t* p = buf + r * stride + k0 + sel * 8;
  v8h a0 = *(const v8h*)p;
  v8h a1 = *(const v8h*)(p + 16);
  return __builtin_shufflevector(a0, a1, 0,1,2,3,4,5,6,7,8,9,10,11,12,13,14,15);
}
// Same A pattern but from GLOBAL f16 (stride HDIM), per-lane row index supplied by caller.
__device__ __forceinline__ v16h load_fragA_gl(const half_t* __restrict__ g, long long row, int k0, int sel) {
  const half_t* p = g + row * HDIM + k0 + sel * 8;
  v8h a0 = *(const v8h*)p;
  v8h a1 = *(const v8h*)(p + 16);
  return __builtin_shufflevector(a0, a1, 0,1,2,3,4,5,6,7,8,9,10,11,12,13,14,15);
}
// B 32x16 from transposed weights Wt[n][k]: lane -> col N=lane%16; lane<16 K {0..7,16..23}, lane>=16 K {8..15,24..31}
__device__ __forceinline__ v16h load_fragB(const half_t* wT, int stride, int n0, int k0, int lane) {
  const int sel = lane >> 4;
  const int n   = n0 + (lane & 15);
  const half_t* p = wT + n * stride + k0 + sel * 8;
  v8h b0 = *(const v8h*)p;
  v8h b1 = *(const v8h*)(p + 16);
  return __builtin_shufflevector(b0, b1, 0,1,2,3,4,5,6,7,8,9,10,11,12,13,14,15);
}
// e1 input A-fragment built straight from global f32 obs (K=32: 24 real dims + 8 zero pad).
__device__ __forceinline__ v16h load_fragA_in(const float* __restrict__ self_obs,
                                              const float* __restrict__ obs,
                                              long long r0g, int row0, int lane, int Bn) {
  const int sel = lane >> 4;
  const long long gr = r0g + row0 + (lane & 15);
  const long long si = gr % (long long)Bn;              // jnp.tile(self_obs,(K,1)) semantics
  const float* sp = self_obs + si * SELF_DIM;
  const float* np = obs + (gr / KNBR) * OBS_DIM + SELF_DIM + (long long)(int)(gr % KNBR) * NBR_DIM;
  v16h f;
#pragma unroll
  for (int e = 0; e < 8; ++e) f[e] = (half_t)sp[sel * 8 + e];   // K = sel*8+e in [0,16): always self
  if (sel == 0) {           // K = 16..23 : self[16..17] + nbr[0..5]
    f[8] = (half_t)sp[16];
    f[9] = (half_t)sp[17];
#pragma unroll
    for (int e = 0; e < 6; ++e) f[10 + e] = (half_t)np[e];
  } else {                  // K = 24..31 : zero pad
#pragma unroll
    for (int e = 8; e < 16; ++e) f[e] = (half_t)0.0f;
  }
  return f;
}

// ---- bulk f16 copy: prepped weights (global) -> LDS, b128 granularity ----
__device__ __forceinline__ void copy_w(const half_t* __restrict__ src, half_t* dst,
                                       int halves, int tid) {
  const v8h* s = (const v8h*)src;
  v8h* d = (v8h*)dst;
  for (int i = tid; i < halves / 8; i += NTHREADS) d[i] = s[i];
}

// ---- WMMA core: out = tanh(A @ W + b) for one 16-row tile, A-fragments in registers ----
template <int KCHUNKS>
__device__ __forceinline__ void mlp_core(const v16h* af, const half_t* Wt, int wstride,
                                         const float* bias, half_t* outL, int row0, int lane) {
  const int sel = lane >> 4;
  const int nlo = lane & 15;
#pragma unroll 2
  for (int nt = 0; nt < 8; ++nt) {
    const int n0 = nt * 16;
    float bv = bias[n0 + nlo];
    v8f acc = {bv, bv, bv, bv, bv, bv, bv, bv};
#pragma unroll
    for (int kc = 0; kc < KCHUNKS; ++kc) {
      v16h bf = load_fragB(Wt, wstride, n0, kc * 32, lane);
      acc = __builtin_amdgcn_wmma_f32_16x16x32_f16(false, af[kc], false, bf, (short)0, acc, false, false);
    }
    const int col = n0 + nlo;
#pragma unroll
    for (int j = 0; j < 8; ++j)
      outL[(row0 + sel * 8 + j) * HDIM + col] = (half_t)ftanh(acc[j]);
  }
}
// Wrapper: A staged in LDS (row-major f16, stride astride)
template <int KCHUNKS>
__device__ __forceinline__ void mlp_layer(const half_t* A, int astride,
                                          const half_t* Wt, int wstride,
                                          const float* bias, half_t* outL,
                                          int row0, int lane) {
  v16h af[KCHUNKS];
#pragma unroll
  for (int kc = 0; kc < KCHUNKS; ++kc) af[kc] = load_fragA(A, astride, row0, kc * 32, lane);
  mlp_core<KCHUNKS>(af, Wt, wstride, bias, outL, row0, lane);
}

// ============ Prep kernel: W (Kd x 128 f32) -> Wt[128][Kpad] f16 (run once) ============
__global__ void prep_kernel(const float* __restrict__ W, half_t* __restrict__ Wt,
                            int Kd, int Kpad) {
  int idx = blockIdx.x * blockDim.x + threadIdx.x;
  if (idx >= HDIM * Kpad) return;
  int kk = idx / HDIM;
  int n  = idx - kk * HDIM;
  float v = (kk < Kd) ? W[kk * HDIM + n] : 0.0f;
  Wt[n * Kpad + kk] = (half_t)v;
}

// prepped-weight region layout (halves), contiguous in the LDS staging order:
#define P_E1   0          // 128*32  = 4096
#define P_E2   4096       // 16384
#define P_V1   20480      // 16384
#define P_V2   36864      // 16384
#define P_K1H  53248      // total halves staged by kernel 1
#define P_A1   53248      // 32768
#define P_A2   86016      // 16384
#define P_TOT  102400     // total halves
#define P_K2H  49152      // halves staged by kernel 2 (A1+A2)

// ================= Kernel 1: encoder + value MLPs, per-sample mean =================
#define K1_WE1   0            // 8192 B
#define K1_WE2   8192         // 32768
#define K1_WV1   40960        // 32768
#define K1_WV2   73728        // 32768
#define K1_BE1   106496       // 512 each
#define K1_BE2   107008
#define K1_BV1   107520
#define K1_BV2   108032
#define K1_EMB   108544       // 384*128*2 = 98304
#define K1_HBUF  206848       // 98304
#define SMEM1    305152       // <= 320KB WGP

__global__ __launch_bounds__(NTHREADS) void enc_kernel(
    const float* __restrict__ self_obs, const float* __restrict__ obs,
    const half_t* __restrict__ wprep,
    const float* __restrict__ e_b1, const float* __restrict__ e_b2,
    const float* __restrict__ v_b1, const float* __restrict__ v_b2,
    half_t* __restrict__ gemb, half_t* __restrict__ gval, half_t* __restrict__ gmean,
    int Bn) {
  extern __shared__ char smem[];
  half_t* wE1 = (half_t*)(smem + K1_WE1);
  float*  bE1 = (float*)(smem + K1_BE1);
  float*  bE2 = (float*)(smem + K1_BE2);
  float*  bV1 = (float*)(smem + K1_BV1);
  float*  bV2 = (float*)(smem + K1_BV2);
  half_t* emb = (half_t*)(smem + K1_EMB);
  half_t* hbf = (half_t*)(smem + K1_HBUF);
  half_t* wE2 = (half_t*)(smem + K1_WE2);
  half_t* wV1 = (half_t*)(smem + K1_WV1);
  half_t* wV2 = (half_t*)(smem + K1_WV2);

  const int tid  = threadIdx.x;
  const int lane = tid & 31;
  const int wave = tid >> 5;
  const int row0 = wave * 16;
  const long long s0  = (long long)blockIdx.x * SPB;
  const long long r0g = s0 * KNBR;

  // e1 input fragment straight from global (no LDS staging)
  v16h afin = load_fragA_in(self_obs, obs, r0g, row0, lane, Bn);

  // stage all 4 weight matrices with one flat b128 copy (prepped layout == LDS layout)
  copy_w(wprep + P_E1, wE1, P_K1H, tid);
  for (int i = tid; i < HDIM; i += NTHREADS) {
    bE1[i] = e_b1[i]; bE2[i] = e_b2[i]; bV1[i] = v_b1[i]; bV2[i] = v_b2[i];
  }
  __syncthreads();

  mlp_core<1>(&afin, wE1, IN1PAD, bE1, hbf, row0, lane);       // e1 -> hbf
  mlp_layer<4>(hbf, HDIM, wE2, HDIM, bE2, emb, row0, lane);    // e2 -> emb
  __syncthreads();

  // export embeddings (b128 coalesced) + per-sample mean
  {
    const v8h* es = (const v8h*)emb;
    v8h* gd = (v8h*)(gemb + r0g * HDIM);
    for (int i = tid; i < RPB * HDIM / 8; i += NTHREADS) gd[i] = es[i];
  }
  for (int i = tid; i < SPB * HDIM / 8; i += NTHREADS) {
    int s = i >> 4, cb = (i & 15) * 8;
    float o[8] = {0, 0, 0, 0, 0, 0, 0, 0};
#pragma unroll
    for (int k = 0; k < KNBR; ++k) {
      v8h e = *(const v8h*)(emb + (s * KNBR + k) * HDIM + cb);
#pragma unroll
      for (int j = 0; j < 8; ++j) o[j] += (float)e[j];
    }
    v8h r;
#pragma unroll
    for (int j = 0; j < 8; ++j) r[j] = (half_t)(o[j] * (1.0f / 6.0f));
    *(v8h*)(gmean + (s0 + s) * HDIM + cb) = r;
  }
  __syncthreads();   // exports done before v2 overwrites emb

  mlp_layer<4>(emb, HDIM, wV1, HDIM, bV1, hbf, row0, lane);    // v1
  mlp_layer<4>(hbf, HDIM, wV2, HDIM, bV2, emb, row0, lane);    // v2 -> emb (scratch)
  __syncthreads();

  {
    const v8h* vs = (const v8h*)emb;
    v8h* gd = (v8h*)(gval + r0g * HDIM);
    for (int i = tid; i < RPB * HDIM / 8; i += NTHREADS) gd[i] = vs[i];
  }
}

// ================= Kernel 2: attention MLP + softmax + weighted sum =================
#define K2_WA1   0            // 65536
#define K2_WA2   65536        // 32768
#define K2_BA1   98304        // 512
#define K2_BA2   98816        // 512
#define K2_WA3   99328        // 512 (f32)
#define K2_HBUF  99840        // 98304
#define K2_H2    198144       // 98304
#define K2_LOG   296448       // 1536
#define K2_ATT   297984       // 1536
#define SMEM2    299520

__global__ __launch_bounds__(NTHREADS) void att_kernel(
    const half_t* __restrict__ gemb, const half_t* __restrict__ gval,
    const half_t* __restrict__ gmean, const half_t* __restrict__ wprep,
    const float* __restrict__ a_b1, const float* __restrict__ a_b2,
    const float* __restrict__ a_w3, const float* __restrict__ a_b3,
    float* __restrict__ out, int Bn) {
  extern __shared__ char smem[];
  half_t* wA1  = (half_t*)(smem + K2_WA1);   // [128][256]
  half_t* wA2  = (half_t*)(smem + K2_WA2);
  float*  bA1  = (float*)(smem + K2_BA1);
  float*  bA2  = (float*)(smem + K2_BA2);
  float*  wA3f = (float*)(smem + K2_WA3);
  half_t* hbf  = (half_t*)(smem + K2_HBUF);
  half_t* h2   = (half_t*)(smem + K2_H2);
  float*  logi = (float*)(smem + K2_LOG);
  float*  attw = (float*)(smem + K2_ATT);

  const int tid  = threadIdx.x;
  const int lane = tid & 31;
  const int wave = tid >> 5;
  const int row0 = wave * 16;
  const int sel  = lane >> 4;
  const long long s0  = (long long)blockIdx.x * SPB;
  const long long r0g = s0 * KNBR;
  const float ab3 = a_b3[0];

  // a1 A-fragments straight from global: af[0..3] = emb rows, af[4..7] = mean[(r) % B] rows
  v16h af[8];
  {
    const long long rowe = r0g + row0 + (lane & 15);
    const long long rowm = rowe % (long long)Bn;       // jnp.tile(emb_mean,(K,1)) semantics
#pragma unroll
    for (int kc = 0; kc < 4; ++kc) {
      af[kc]     = load_fragA_gl(gemb,  rowe, kc * 32, sel);
      af[4 + kc] = load_fragA_gl(gmean, rowm, kc * 32, sel);
    }
  }

  copy_w(wprep + P_A1, wA1, P_K2H, tid);   // a1 (k 0..127 emb, 128..255 mean) + a2, flat copy
  for (int i = tid; i < HDIM; i += NTHREADS) {
    bA1[i] = a_b1[i]; bA2[i] = a_b2[i]; wA3f[i] = a_w3[i];
  }
  __syncthreads();

  mlp_core<8>(af, wA1, 256, bA1, hbf, row0, lane);             // a1 -> hbf
  mlp_layer<4>(hbf, HDIM, wA2, HDIM, bA2, h2, row0, lane);     // a2 -> h2
  // logits: one lane per row, b128 LDS reads
  if (lane < 16) {
    int r = row0 + lane;
    float acc = ab3;
    const v8h* pr = (const v8h*)(h2 + r * HDIM);
    for (int cb = 0; cb < 16; ++cb) {
      v8h hv = pr[cb];
#pragma unroll
      for (int j = 0; j < 8; ++j) acc += (float)hv[j] * wA3f[cb * 8 + j];
    }
    logi[r] = acc;
  }
  __syncthreads();
  // softmax over 6 logits per sample
  if (tid < SPB) {
    float m = -3.0e38f;
    float e[KNBR];
#pragma unroll
    for (int k = 0; k < KNBR; ++k) m = fmaxf(m, logi[tid * KNBR + k]);
    float s = 0.0f;
#pragma unroll
    for (int k = 0; k < KNBR; ++k) { e[k] = __expf(logi[tid * KNBR + k] - m); s += e[k]; }
    float rs = __builtin_amdgcn_rcpf(s);
#pragma unroll
    for (int k = 0; k < KNBR; ++k) attw[tid * KNBR + k] = e[k] * rs;
  }
  __syncthreads();
  // out[s][cb..cb+7] = sum_k att[s,k] * val[s*6+k][cb..cb+7]  (b128 loads, v8f store)
  for (int i = tid; i < SPB * HDIM / 8; i += NTHREADS) {
    int s = i >> 4, cb = (i & 15) * 8;
    float o[8] = {0, 0, 0, 0, 0, 0, 0, 0};
#pragma unroll
    for (int k = 0; k < KNBR; ++k) {
      float w = attw[s * KNBR + k];
      v8h vv = *(const v8h*)(gval + (r0g + s * KNBR + k) * (long long)HDIM + cb);
#pragma unroll
      for (int j = 0; j < 8; ++j) o[j] += w * (float)vv[j];
    }
    v8f ov;
#pragma unroll
    for (int j = 0; j < 8; ++j) ov[j] = o[j];
    *(v8f*)(out + (s0 + s) * (long long)HDIM + cb) = ov;
  }
}

extern "C" void kernel_launch(void* const* d_in, const int* in_sizes, int n_in,
                              void* d_out, int out_size, void* d_ws, size_t ws_size,
                              hipStream_t stream) {
  const float* self_obs = (const float*)d_in[0];
  const float* obs      = (const float*)d_in[1];
  const float* e_w1 = (const float*)d_in[2];
  const float* e_b1 = (const float*)d_in[3];
  const float* e_w2 = (const float*)d_in[4];
  const float* e_b2 = (const float*)d_in[5];
  const float* v_w1 = (const float*)d_in[6];
  const float* v_b1 = (const float*)d_in[7];
  const float* v_w2 = (const float*)d_in[8];
  const float* v_b2 = (const float*)d_in[9];
  const float* a_w1 = (const float*)d_in[10];
  const float* a_b1 = (const float*)d_in[11];
  const float* a_w2 = (const float*)d_in[12];
  const float* a_b2 = (const float*)d_in[13];
  const float* a_w3 = (const float*)d_in[14];
  const float* a_b3 = (const float*)d_in[15];
  float* out = (float*)d_out;

  const int Bn = in_sizes[0] / SELF_DIM;       // 65536
  const long long nrows = (long long)Bn * KNBR;

  char* ws = (char*)d_ws;
  half_t* gemb  = (half_t*)ws;                                   // nrows*128 f16
  half_t* gval  = (half_t*)(ws + nrows * HDIM * 2);              // nrows*128 f16
  half_t* gmean = (half_t*)(ws + 2 * nrows * HDIM * 2);          // Bn*128 f16
  half_t* wprep = (half_t*)(ws + 2 * nrows * HDIM * 2 + (long long)Bn * HDIM * 2);

  // one-time weight prep (tiny): f32 row-major -> transposed f16 fragment layout
  prep_kernel<<<(HDIM * IN1PAD + 255) / 256, 256, 0, stream>>>(e_w1, wprep + P_E1, 24, IN1PAD);
  prep_kernel<<<(HDIM * HDIM + 255) / 256, 256, 0, stream>>>(e_w2, wprep + P_E2, HDIM, HDIM);
  prep_kernel<<<(HDIM * HDIM + 255) / 256, 256, 0, stream>>>(v_w1, wprep + P_V1, HDIM, HDIM);
  prep_kernel<<<(HDIM * HDIM + 255) / 256, 256, 0, stream>>>(v_w2, wprep + P_V2, HDIM, HDIM);
  prep_kernel<<<(HDIM * 256 + 255) / 256, 256, 0, stream>>>(a_w1, wprep + P_A1, 256, 256);
  prep_kernel<<<(HDIM * HDIM + 255) / 256, 256, 0, stream>>>(a_w2, wprep + P_A2, HDIM, HDIM);

  (void)hipFuncSetAttribute((const void*)enc_kernel,
                            hipFuncAttributeMaxDynamicSharedMemorySize, SMEM1);
  (void)hipFuncSetAttribute((const void*)att_kernel,
                            hipFuncAttributeMaxDynamicSharedMemorySize, SMEM2);

  const int grid = Bn / SPB;                   // 1024 blocks
  enc_kernel<<<grid, NTHREADS, SMEM1, stream>>>(self_obs, obs, wprep,
      e_b1, e_b2, v_b1, v_b2, gemb, gval, gmean, Bn);
  att_kernel<<<grid, NTHREADS, SMEM2, stream>>>(gemb, gval, gmean, wprep,
      a_b1, a_b2, a_w3, a_b3, out, Bn);
}
